// EntityDisambiguationHead_66142496358617
// MI455X (gfx1250) — matrix operand order repaired and checked by reading, not compiled, not run on hardware
//
#include <hip/hip_runtime.h>
#include <hip/hip_bf16.h>

typedef __attribute__((ext_vector_type(8)))  _Float16 v8h;
typedef __attribute__((ext_vector_type(16))) _Float16 v16h;
typedef __attribute__((ext_vector_type(8)))  float    v8f;

#define DIN    768
#define DE     256
#define M_ROWS 512      // B*S = 4*128
#define E_ROWS 100000
#define EPS_F  1e-8f

// ---------------------------------------------------------------------------
// Kernel 1: q = tanh(x @ W^T + b); row-normalize; emit f16 qn [512][256].
// ---------------------------------------------------------------------------
__global__ __launch_bounds__(256) void k_proj(const float* __restrict__ x,
                                              const float* __restrict__ W,
                                              const float* __restrict__ bias,
                                              _Float16* __restrict__ qn) {
  __shared__ float xs[8][DIN];
  __shared__ float qs[8][DE];
  __shared__ float norms[8];
  const int t = threadIdx.x;
  const int row0 = blockIdx.x * 8;

  for (int i = t; i < 8 * DIN; i += 256) {
    xs[i / DIN][i % DIN] = x[(size_t)(row0 + i / DIN) * DIN + (i % DIN)];
  }
  __syncthreads();

  float acc[8] = {0.f, 0.f, 0.f, 0.f, 0.f, 0.f, 0.f, 0.f};
  const float* wr = W + (size_t)t * DIN;
  for (int i = 0; i < DIN; ++i) {
    const float w = wr[i];
#pragma unroll
    for (int r = 0; r < 8; ++r) acc[r] += xs[r][i] * w;
  }
  const float bb = bias[t];
#pragma unroll
  for (int r = 0; r < 8; ++r) qs[r][t] = tanhf(acc[r] + bb);
  __syncthreads();

  const int r = t >> 5, lane = t & 31;
  float s = 0.f;
  for (int i = lane; i < DE; i += 32) { const float v = qs[r][i]; s += v * v; }
#pragma unroll
  for (int off = 16; off > 0; off >>= 1) s += __shfl_xor(s, off, 32);
  if (lane == 0) norms[r] = s;
  __syncthreads();

#pragma unroll
  for (int rr = 0; rr < 8; ++rr) {
    const float sc = 1.f / fmaxf(sqrtf(norms[rr]), EPS_F);
    qn[(size_t)(row0 + rr) * DE + t] = (_Float16)(qs[rr][t] * sc);
  }
}

// ---------------------------------------------------------------------------
// Kernel 2: L2-normalize entity embeddings, convert to f16 (kept cacheable:
// it is re-read by the GEMM and fits in the 192MB L2).
// ---------------------------------------------------------------------------
__global__ __launch_bounds__(256) void k_norm_e(const float* __restrict__ ee,
                                                _Float16* __restrict__ en) {
  const int warp = (int)((blockIdx.x * (unsigned)blockDim.x + threadIdx.x) >> 5);
  const int lane = threadIdx.x & 31;
  if (warp >= E_ROWS) return;

  const float* src = ee + (size_t)warp * DE + lane * 8;
  float v[8];
#pragma unroll
  for (int i = 0; i < 8; ++i) v[i] = src[i];
  float s = 0.f;
#pragma unroll
  for (int i = 0; i < 8; ++i) s += v[i] * v[i];
#pragma unroll
  for (int off = 16; off > 0; off >>= 1) s += __shfl_xor(s, off, 32);
  const float sc = 1.f / fmaxf(sqrtf(s), EPS_F);

  v8h h;
#pragma unroll
  for (int i = 0; i < 8; ++i) h[i] = (_Float16)(v[i] * sc);
  *(v8h*)(en + (size_t)warp * DE + lane * 8) = h;
}

// ---------------------------------------------------------------------------
// Kernel 3 helpers
// ---------------------------------------------------------------------------
__device__ __forceinline__ v8f wmma_f16(const v16h a, const v16h b, const v8f c) {
  return __builtin_amdgcn_wmma_f32_16x16x32_f16(false, a, false, b, (short)0, c,
                                                false, false);
}

// Load a full 16x256 A tile for one lane: 16 b128 loads issued as one clause.
// A 16x32 f16, contiguous k order used here (same as B): lane M = lane%16; lanes 0-15 hold K 0-15 (v0-7);
// lanes 16-31 hold K 16-31  (ap already includes hi*16).
__device__ __forceinline__ void load_atile(const _Float16* __restrict__ ap,
                                           v16h (&A)[8]) {
#pragma unroll
  for (int ks = 0; ks < 8; ++ks) {
    const v8h lo = *(const v8h*)(ap + ks * 32);
    const v8h hh = *(const v8h*)(ap + ks * 32 + 8);
    A[ks] = __builtin_shufflevector(lo, hh, 0, 1, 2, 3, 4, 5, 6, 7, 8, 9, 10,
                                    11, 12, 13, 14, 15);
  }
}

// D store: VGPR r -> M = m_base + r + 8*hi, N = e_base + (nt?16:0) + n.
// Non-temporal: out is 205MB write-once, keep it out of L2.
__device__ __forceinline__ void store_tile(float* __restrict__ out, int m_base,
                                           int hi, int n, int e_base,
                                           const v8f acc0, const v8f acc1) {
  float* o0 = out + (size_t)(m_base + hi * 8) * E_ROWS + e_base + n;
#pragma unroll
  for (int rr = 0; rr < 8; ++rr) {
    __builtin_nontemporal_store(acc0[rr], o0 + (size_t)rr * E_ROWS);
    __builtin_nontemporal_store(acc1[rr], o0 + (size_t)rr * E_ROWS + 16);
  }
}

// ---------------------------------------------------------------------------
// Kernel 3: out[512,100000] = qn @ en^T via v_wmma_f32_16x16x32_f16.
// One wave per job = 2 adjacent N-tiles (32 entities). The full 32x256 f16
// B-panel lives in 128 VGPRs (fetched from memory exactly once). A tiles are
// double-buffered across M-tiles: a 16-load clause for tile mt+1 is in
// flight while the 16 WMMAs of tile mt execute, so waits are partial
// (loadcnt<=16) instead of full stops.
//
// B 32x16 f16, contiguous k order used here (same as A): lane N = lane%16; lanes 0-15 hold K 0-15 (v0-7);
// lanes 16-31 hold K 16-31.
// ---------------------------------------------------------------------------
__global__ __launch_bounds__(256) void k_cosgemm(const _Float16* __restrict__ qn,
                                                 const _Float16* __restrict__ en,
                                                 float* __restrict__ out) {
  const int NJOBS = (E_ROWS / 16) / 2;              // 3125
  const int wid = threadIdx.x >> 5;
  const int job = blockIdx.x * 8 + wid;
  if (job >= NJOBS) return;                         // wave-uniform exit

  const int lane = threadIdx.x & 31;
  const int n  = lane & 15;
  const int hi = lane >> 4;
  const int e_base = job * 32;

  // Preload B: 2 N-tiles x 8 K-steps -> 128 VGPRs.
  v16h B0[8], B1[8];
  {
    const _Float16* p0 = en + (size_t)(e_base + n) * DE + hi * 16;
    const _Float16* p1 = p0 + (size_t)16 * DE;
#pragma unroll
    for (int ks = 0; ks < 8; ++ks) {
      v8h lo = *(const v8h*)(p0 + ks * 32);
      v8h hh = *(const v8h*)(p0 + ks * 32 + 8);
      B0[ks] = __builtin_shufflevector(lo, hh, 0, 1, 2, 3, 4, 5, 6, 7, 8, 9,
                                       10, 11, 12, 13, 14, 15);
      lo = *(const v8h*)(p1 + ks * 32);
      hh = *(const v8h*)(p1 + ks * 32 + 8);
      B1[ks] = __builtin_shufflevector(lo, hh, 0, 1, 2, 3, 4, 5, 6, 7, 8, 9,
                                       10, 11, 12, 13, 14, 15);
    }
  }

  const _Float16* aq = qn + (size_t)n * DE + hi * 16;

  v16h Aa[8], Ab[8];
  load_atile(aq, Aa);                               // prologue: tile mt=0

#pragma unroll 1
  for (int mt = 0; mt < 32; mt += 2) {
    // Issue loads for tile mt+1 while computing tile mt.
    load_atile(aq + (size_t)(mt + 1) * 16 * DE, Ab);
    v8f acc0 = {}, acc1 = {};
#pragma unroll
    for (int ks = 0; ks < 8; ++ks) {
      acc0 = wmma_f16(Aa[ks], B0[ks], acc0);
      acc1 = wmma_f16(Aa[ks], B1[ks], acc1);
    }
    store_tile(out, mt * 16, hi, n, e_base, acc0, acc1);

    // Issue loads for tile mt+2 while computing tile mt+1.
    if (mt + 2 < 32) load_atile(aq + (size_t)(mt + 2) * 16 * DE, Aa);
    v8f acc2 = {}, acc3 = {};
#pragma unroll
    for (int ks = 0; ks < 8; ++ks) {
      acc2 = wmma_f16(Ab[ks], B0[ks], acc2);
      acc3 = wmma_f16(Ab[ks], B1[ks], acc3);
    }
    store_tile(out, (mt + 1) * 16, hi, n, e_base, acc2, acc3);
  }
}

// ---------------------------------------------------------------------------
extern "C" void kernel_launch(void* const* d_in, const int* in_sizes, int n_in,
                              void* d_out, int out_size, void* d_ws, size_t ws_size,
                              hipStream_t stream) {
  (void)in_sizes; (void)n_in; (void)out_size; (void)ws_size;
  const float* x  = (const float*)d_in[0];   // [4,128,768]
  const float* W  = (const float*)d_in[1];   // [256,768]
  const float* b  = (const float*)d_in[2];   // [256]
  const float* ee = (const float*)d_in[3];   // [100000,256]
  float* out = (float*)d_out;                // [4,128,100000]

  _Float16* qn = (_Float16*)d_ws;                                        // 512*256 f16
  _Float16* en = (_Float16*)((char*)d_ws + (size_t)M_ROWS * DE * 2);     // 100000*256 f16

  k_proj   <<<M_ROWS / 8,               256, 0, stream>>>(x, W, b, qn);
  k_norm_e <<<(E_ROWS + 7) / 8,         256, 0, stream>>>(ee, en);
  k_cosgemm<<<((E_ROWS / 32) + 7) / 8,  256, 0, stream>>>(qn, en, out);
}
